// CausalSelfAttention_6897717477730
// MI455X (gfx1250) — matrix operand order repaired and checked by reading, not compiled, not run on hardware
//
#include <hip/hip_runtime.h>
#include <hip/hip_bf16.h>

typedef __bf16 bf16;
typedef __bf16 v8bf  __attribute__((ext_vector_type(8)));
typedef __bf16 v16bf __attribute__((ext_vector_type(16)));
typedef float  v8f   __attribute__((ext_vector_type(8)));
typedef unsigned int v4u __attribute__((ext_vector_type(4)));
typedef int    v8i   __attribute__((ext_vector_type(8)));
typedef int    v4i   __attribute__((ext_vector_type(4)));

static __device__ __forceinline__ v8f zero_v8f() {
  v8f z;
#pragma unroll
  for (int i = 0; i < 8; ++i) z[i] = 0.0f;
  return z;
}

static __device__ __forceinline__ v16bf join16(v8bf a, v8bf b) {
  return __builtin_shufflevector(a, b, 0,1,2,3,4,5,6,7,8,9,10,11,12,13,14,15);
}

static __device__ __forceinline__ v8f wmma_bf16(v16bf a, v16bf b, v8f c) {
  // D = A(16x32 bf16) * B(32x16 bf16) + C(16x16 f32)
  return __builtin_amdgcn_wmma_f32_16x16x32_bf16(false, a, false, b, (short)0, c, false, false);
}

// ---------------------------------------------------------------------------
// Tensor Data Mover: 2D bf16 tile (tile_d0 fastest/contig, tile_d1 rows) from
// global row-major (row pitch in elements) into LDS, rows packed contiguously.
// ---------------------------------------------------------------------------
static __device__ __forceinline__ void tdm_load_tile_bf16(
    unsigned lds_off, const bf16* gsrc,
    unsigned tile_d0, unsigned tile_d1, unsigned long long row_pitch_elems)
{
  v4u g0;
  unsigned long long ga = (unsigned long long)gsrc;
  g0[0] = 1u;                                               // count=1, user mode
  g0[1] = lds_off;                                          // LDS byte address
  g0[2] = (unsigned)(ga & 0xffffffffull);                   // global_addr[31:0]
  g0[3] = (unsigned)((ga >> 32) & 0x1ffffffull) | (2u << 30); // addr[56:32], type=2

  v8i g1;
  g1[0] = (int)(1u << 16);                                  // data_size=1 (2 bytes)
  g1[1] = (int)((tile_d0 & 0xffffu) << 16);                 // tensor_dim0 lo16
  g1[2] = (int)(((tile_d0 >> 16) & 0xffffu) | ((tile_d1 & 0xffffu) << 16)); // dim0 hi / dim1 lo
  g1[3] = (int)(((tile_d1 >> 16) & 0xffffu) | ((tile_d0 & 0xffffu) << 16)); // dim1 hi / tile_dim0
  g1[4] = (int)(tile_d1 & 0xffffu);                         // tile_dim1 ; tile_dim2=0
  g1[5] = (int)(unsigned)(row_pitch_elems & 0xffffffffull); // tensor_dim0_stride lo32
  g1[6] = (int)(unsigned)((row_pitch_elems >> 32) & 0xffffull); // stride hi16 ; dim1_stride lo16=0
  g1[7] = 0;

  v4i z4 = {0, 0, 0, 0};
#if __has_include(<hip/amd_detail/amd_gfx1250_TDM.h>)
  v8i z8 = {0, 0, 0, 0, 0, 0, 0, 0};
  __builtin_amdgcn_tensor_load_to_lds(g0, g1, z4, z4, z8, 0);
#else
  __builtin_amdgcn_tensor_load_to_lds(g0, g1, z4, z4, 0);
#endif
}

// ---------------------------------------------------------------------------
// Prep kernels: fp32 -> bf16 cast, and cast+transpose for weights.
// ---------------------------------------------------------------------------
__global__ void cast_bf16_kernel(const float* __restrict__ s, bf16* __restrict__ d, int n) {
  int i = blockIdx.x * blockDim.x + threadIdx.x;
  if (i < n) d[i] = (bf16)s[i];
}

// src [R][C] f32 -> dst [C][R] bf16
__global__ void transpose_cast_kernel(const float* __restrict__ src, bf16* __restrict__ dst,
                                      int R, int C) {
  __shared__ float tile[32][33];
  const int c0 = blockIdx.x * 32, r0 = blockIdx.y * 32;
  const int tx = threadIdx.x & 31, ty = threadIdx.x >> 5;   // 256 threads: ty 0..7
#pragma unroll
  for (int i = ty; i < 32; i += 8)
    tile[i][tx] = src[(size_t)(r0 + i) * C + c0 + tx];
  __syncthreads();
#pragma unroll
  for (int i = ty; i < 32; i += 8)
    dst[(size_t)(c0 + i) * R + r0 + tx] = (bf16)tile[tx][i];
}

// ---------------------------------------------------------------------------
// Tiled GEMM:  A[M][1024] (bf16, row-major) x BT[N][1024] (bf16, B transposed)
// Block tile 128(M) x 64(N), 8 waves, K staged via TDM double-buffered LDS.
// MODE 0: epilogue scatters qkv -> q/k row-major per head, v transposed.
// MODE 1: epilogue writes f32 out + bias.
// ---------------------------------------------------------------------------
template <int MODE>
__global__ __launch_bounds__(256) void gemm_bf16_kernel(
    const bf16* __restrict__ A, const bf16* __restrict__ BT,
    const float* __restrict__ bias,
    bf16* __restrict__ qb, bf16* __restrict__ kb, bf16* __restrict__ vT,
    float* __restrict__ outF)
{
  __shared__ __align__(16) bf16 Al[2][128 * 32];
  __shared__ __align__(16) bf16 Bl[2][64 * 32];

  const int M0 = blockIdx.x * 128;
  const int N0 = blockIdx.y * 64;
  const int tid = threadIdx.x;
  const int w = tid >> 5, l = tid & 31, lc = l & 15, half = l >> 4;
  const bool leader = (tid < 32);

  constexpr int KTOT = 1024, KT = 32;
  const int nk = KTOT / KT;

  if (leader) {
    tdm_load_tile_bf16((unsigned)(unsigned long long)&Al[0][0],
                       A + (size_t)M0 * KTOT, KT, 128, KTOT);
    tdm_load_tile_bf16((unsigned)(unsigned long long)&Bl[0][0],
                       BT + (size_t)N0 * KTOT, KT, 64, KTOT);
  }

  v8f acc[4];
#pragma unroll
  for (int j = 0; j < 4; ++j) acc[j] = zero_v8f();

  for (int it = 0; it < nk; ++it) {
    const int cur = it & 1, nxt = cur ^ 1;
    if (leader) {
      if (it + 1 < nk) {
        const int k1 = (it + 1) * KT;
        tdm_load_tile_bf16((unsigned)(unsigned long long)&Al[nxt][0],
                           A + (size_t)M0 * KTOT + k1, KT, 128, KTOT);
        tdm_load_tile_bf16((unsigned)(unsigned long long)&Bl[nxt][0],
                           BT + (size_t)N0 * KTOT + k1, KT, 64, KTOT);
        __builtin_amdgcn_s_wait_tensorcnt(2);   // current buffer's 2 loads done
      } else {
        __builtin_amdgcn_s_wait_tensorcnt(0);
      }
    }
    __syncthreads();

    const bf16* a_ = &Al[cur][(16 * w + lc) * 32];
    v16bf av = join16(*(const v8bf*)(a_ + half * 8),
                      *(const v8bf*)(a_ + 16 + half * 8));
#pragma unroll
    for (int j = 0; j < 4; ++j) {
      const bf16* b_ = &Bl[cur][(16 * j + lc) * 32];
      v16bf bv = join16(*(const v8bf*)(b_ + half * 8), *(const v8bf*)(b_ + 16 + half * 8));
      acc[j] = wmma_bf16(av, bv, acc[j]);
    }
    __syncthreads();   // everyone done reading buf[cur] before it is refilled
  }

  // epilogue
#pragma unroll
  for (int j = 0; j < 4; ++j) {
    const int n = N0 + 16 * j + lc;
    const float bn = bias[n];
#pragma unroll
    for (int r = 0; r < 8; ++r) {
      const int m = half ? r + 8 : r;
      const int t = M0 + 16 * w + m;
      const float v = acc[j][r] + bn;
      if (MODE == 0) {
        const int which = n >> 10;
        const int c = n & 1023;
        const int h = c >> 6, d = c & 63;
        const int b = t >> 11, tt = t & 2047;
        const int bh = b * 16 + h;
        if (which == 0)      qb[((size_t)(bh * 2048 + tt)) * 64 + d] = (bf16)v;
        else if (which == 1) kb[((size_t)(bh * 2048 + tt)) * 64 + d] = (bf16)v;
        else                 vT[((size_t)(bh * 64 + d)) * 2048 + tt] = (bf16)v;
      } else {
        outF[(size_t)t * 1024 + n] = v;
      }
    }
  }
}

// ---------------------------------------------------------------------------
// Flash attention: block = (b,h,64-row q tile), 4 waves x 16 q rows.
// q,k row-major [T][64] bf16; v transposed [64][T] bf16; y -> [B*T][1024] bf16.
// ---------------------------------------------------------------------------
__global__ __launch_bounds__(128) void attn_kernel(
    const bf16* __restrict__ qb, const bf16* __restrict__ kb,
    const bf16* __restrict__ vT, bf16* __restrict__ yb)
{
  __shared__ __align__(16) bf16 Plds[4][16 * 32];

  const int qt = blockIdx.x & 31;
  const int bh = blockIdx.x >> 5;
  const int b = bh >> 4, h = bh & 15;
  const int Q0 = qt * 64;
  const int tid = threadIdx.x;
  const int w = tid >> 5, l = tid & 31, lc = l & 15, half = l >> 4;
  const int qr = Q0 + 16 * w;

  const bf16* qh = qb + (size_t)bh * 2048 * 64;
  const bf16* kh = kb + (size_t)bh * 2048 * 64;
  const bf16* vh = vT + (size_t)bh * 64 * 2048;

  // q tile A-operands, reused over the whole kv loop (2 K-steps of 32 over hs=64)
  v16bf qa[2];
#pragma unroll
  for (int kd = 0; kd < 2; ++kd) {
    const bf16* p = qh + (size_t)(qr + lc) * 64 + kd * 32;
    qa[kd] = join16(*(const v8bf*)(p + half * 8),
                    *(const v8bf*)(p + 16 + half * 8));
  }

  float m_i[8], l_i[8];
  v8f O[4];
#pragma unroll
  for (int j = 0; j < 4; ++j) O[j] = zero_v8f();
#pragma unroll
  for (int r = 0; r < 8; ++r) { m_i[r] = -__builtin_inff(); l_i[r] = 0.0f; }

  const float scale = 0.125f;          // 1/sqrt(64)
  const int nkv = Q0 / 32 + 2;         // causal: key tiles covering key <= Q0+63
  bf16* pl = &Plds[w][0];

  for (int kt = 0; kt < nkv; ++kt) {
    const int key0 = kt * 32;
    v8f s0 = zero_v8f(), s1 = zero_v8f();
#pragma unroll
    for (int kd = 0; kd < 2; ++kd) {
      const bf16* p0 = kh + (size_t)(key0 + lc) * 64 + kd * 32;
      s0 = wmma_bf16(qa[kd], join16(*(const v8bf*)(p0 + half * 8), *(const v8bf*)(p0 + 16 + half * 8)), s0);
      const bf16* p1 = kh + (size_t)(key0 + 16 + lc) * 64 + kd * 32;
      s1 = wmma_bf16(qa[kd], join16(*(const v8bf*)(p1 + half * 8), *(const v8bf*)(p1 + 16 + half * 8)), s1);
    }

    float alpha[8];
#pragma unroll
    for (int r = 0; r < 8; ++r) {
      const int row = qr + (half ? r + 8 : r);
      const int c0 = key0 + lc, c1 = key0 + 16 + lc;
      float v0 = (c0 <= row) ? s0[r] * scale : -__builtin_inff();
      float v1 = (c1 <= row) ? s1[r] * scale : -__builtin_inff();
      // row max over 32 cols (xor masks <16 stay inside each 16-lane half)
      float mx = fmaxf(v0, v1);
      mx = fmaxf(mx, __shfl_xor(mx, 1));
      mx = fmaxf(mx, __shfl_xor(mx, 2));
      mx = fmaxf(mx, __shfl_xor(mx, 4));
      mx = fmaxf(mx, __shfl_xor(mx, 8));
      const float mnew = fmaxf(m_i[r], mx);
      alpha[r] = __expf(m_i[r] - mnew);
      m_i[r] = mnew;
      const float p0e = __expf(v0 - mnew);
      const float p1e = __expf(v1 - mnew);
      float rs = p0e + p1e;
      rs += __shfl_xor(rs, 1);
      rs += __shfl_xor(rs, 2);
      rs += __shfl_xor(rs, 4);
      rs += __shfl_xor(rs, 8);
      l_i[r] = l_i[r] * alpha[r] + rs;
      const int m = half ? r + 8 : r;
      pl[m * 32 + lc]      = (bf16)p0e;   // C-layout -> LDS (row-major 16x32)
      pl[m * 32 + 16 + lc] = (bf16)p1e;
    }

#pragma unroll
    for (int j = 0; j < 4; ++j)
#pragma unroll
      for (int r = 0; r < 8; ++r) O[j][r] *= alpha[r];

    // reload P in A-operand layout (per-wave LDS, DS ops in-order within wave)
    const bf16* pp = pl + lc * 32;
    v16bf pa = join16(*(const v8bf*)(pp + half * 8),
                      *(const v8bf*)(pp + 16 + half * 8));
#pragma unroll
    for (int j = 0; j < 4; ++j) {
      const bf16* vp = vh + (size_t)(16 * j + lc) * 2048 + key0;
      O[j] = wmma_bf16(pa, join16(*(const v8bf*)(vp + half * 8), *(const v8bf*)(vp + 16 + half * 8)), O[j]);
    }
  }

  // normalize + store y (bf16 row-major [B*T][1024] for the proj GEMM)
#pragma unroll
  for (int r = 0; r < 8; ++r) {
    const float inv = 1.0f / l_i[r];
    const int m = half ? r + 8 : r;
    const int tt = qr + m;
#pragma unroll
    for (int j = 0; j < 4; ++j) {
      yb[((size_t)(b * 2048 + tt)) * 1024 + h * 64 + 16 * j + lc] =
          (bf16)(O[j][r] * inv);
    }
  }
}

// ---------------------------------------------------------------------------
// Launch
// ---------------------------------------------------------------------------
extern "C" void kernel_launch(void* const* d_in, const int* in_sizes, int n_in,
                              void* d_out, int out_size, void* d_ws, size_t ws_size,
                              hipStream_t stream) {
  const float* x      = (const float*)d_in[0];   // [2,2048,1024]
  const float* W_attn = (const float*)d_in[1];   // [1024,3072]
  const float* b_attn = (const float*)d_in[2];   // [3072]
  const float* W_proj = (const float*)d_in[3];   // [1024,1024]
  const float* b_proj = (const float*)d_in[4];   // [1024]
  float* out = (float*)d_out;                    // [2,2048,1024] f32

  char* p = (char*)d_ws;
  bf16* xb  = (bf16*)p; p += (size_t)4096 * 1024 * 2;   // x bf16
  bf16* WaT = (bf16*)p; p += (size_t)3072 * 1024 * 2;   // W_attn^T bf16
  bf16* WpT = (bf16*)p; p += (size_t)1024 * 1024 * 2;   // W_proj^T bf16
  bf16* qb  = (bf16*)p; p += (size_t)32 * 2048 * 64 * 2;
  bf16* kb  = (bf16*)p; p += (size_t)32 * 2048 * 64 * 2;
  bf16* vT  = (bf16*)p; p += (size_t)32 * 64 * 2048 * 2;
  bf16* yb  = (bf16*)p; p += (size_t)4096 * 1024 * 2;

  cast_bf16_kernel<<<(4096 * 1024) / 256, 256, 0, stream>>>(x, xb, 4096 * 1024);
  transpose_cast_kernel<<<dim3(3072 / 32, 1024 / 32), 256, 0, stream>>>(W_attn, WaT, 1024, 3072);
  transpose_cast_kernel<<<dim3(1024 / 32, 1024 / 32), 256, 0, stream>>>(W_proj, WpT, 1024, 1024);

  gemm_bf16_kernel<0><<<dim3(4096 / 128, 3072 / 64), 256, 0, stream>>>(
      xb, WaT, b_attn, qb, kb, vT, nullptr);

  attn_kernel<<<dim3(1024), 128, 0, stream>>>(qb, kb, vT, yb);

  gemm_bf16_kernel<1><<<dim3(4096 / 128, 1024 / 64), 256, 0, stream>>>(
      yb, WpT, b_proj, nullptr, nullptr, nullptr, out);
}
